// BiMamba_54168127537187
// MI455X (gfx1250) — matrix operand order repaired and checked by reading, not compiled, not run on hardware
//
#include <hip/hip_runtime.h>
#include <hip/hip_bf16.h>

// ---------------------------------------------------------------------------
// BiMamba for gfx1250 (MI455X): f16 WMMA GEMMs with TDM (tensor_load_to_lds)
// tile staging + register-resident selective scan.
// ---------------------------------------------------------------------------

typedef _Float16 f16;
typedef __attribute__((ext_vector_type(16))) _Float16     v16h;
typedef __attribute__((ext_vector_type(8)))  float        v8f;
typedef __attribute__((ext_vector_type(4)))  unsigned int u32x4;
typedef __attribute__((ext_vector_type(8)))  unsigned int u32x8;

#define DMODEL 256
#define NSTATE 16
#define DINNER 512
#define DTRANK 16
#define DCONV  4
#define BATCH  8
#define SEQ    2048
#define MTOT   (BATCH * SEQ)   // 16384 rows (b*L + l)

union FragH { v16h h; u32x4 q[2]; };

// ---------------------------------------------------------------------------
// Tensor Data Mover: DMA a 2-D f16 tile (tile_rows x 32 elems, row stride =
// k_stride elems) from global into LDS.  D# per CDNA5 ISA ch.8: group0 holds
// {count, lds_addr, global_addr, type=2}; group1 holds {data_size=2B,
// tensor_dim0=k_stride, tensor_dim1=rows_avail (rows past it read as zero),
// tile_dim0=32, tile_dim1=tile_rows, tensor_dim0_stride=k_stride}.
// 2-D tensor -> 2-group form (VADDR2/3 = NULL).  Tracked by TENSORcnt.
// ---------------------------------------------------------------------------
__device__ __forceinline__ void tdm_load_tile_f16(const f16* gsrc, unsigned ldsOff,
                                                  unsigned rows_avail,
                                                  unsigned tile_rows,
                                                  unsigned k_stride) {
    const unsigned long long ga = (unsigned long long)(uintptr_t)gsrc;
    u32x4 g0;
    g0[0] = 1u;                                    // count=1, user descriptor
    g0[1] = ldsOff;                                // lds_addr (bytes)
    g0[2] = (unsigned)(ga & 0xFFFFFFFFu);          // global_addr[31:0]
    g0[3] = (unsigned)((ga >> 32) & 0x1FFFFFFu)    // global_addr[56:32]
          | (2u << 30);                            // type = 2 ("image")
    u32x8 g1;
    g1[0] = (1u << 16);                            // data_size=1 -> 2 bytes
    g1[1] = (k_stride & 0xFFFFu) << 16;            // tensor_dim0[15:0]
    g1[2] = ((k_stride >> 16) & 0xFFFFu)           // tensor_dim0[31:16]
          | ((rows_avail & 0xFFFFu) << 16);        // tensor_dim1[15:0]
    g1[3] = ((rows_avail >> 16) & 0xFFFFu)         // tensor_dim1[31:16]
          | (32u << 16);                           // tile_dim0 = 32 elems
    g1[4] = tile_rows & 0xFFFFu;                   // tile_dim1 (tile_dim2=0)
    g1[5] = k_stride;                              // tensor_dim0_stride[31:0]
    g1[6] = 0u;                                    // stride0_hi / stride1_lo
    g1[7] = 0u;                                    // stride1_hi
    asm volatile("tensor_load_to_lds %0, %1" :: "s"(g0), "s"(g1) : "memory");
}

// ---------------------------------------------------------------------------
// Kernel 0a: transpose x (B, D, L) -> row-major (B*L, D) f16, fwd + L-flipped.
// ---------------------------------------------------------------------------
__global__ void prep_x(const float* __restrict__ x,
                       f16* __restrict__ xtF, f16* __restrict__ xtB) {
    int idx = blockIdx.x * blockDim.x + threadIdx.x;
    if (idx >= BATCH * DMODEL * SEQ) return;
    int l = idx % SEQ;
    int t = idx / SEQ;
    int d = t % DMODEL;
    int b = t / DMODEL;
    float v = x[idx];
    xtF[((size_t)(b * SEQ + l)) * DMODEL + d]            = (f16)v;
    xtB[((size_t)(b * SEQ + (SEQ - 1 - l))) * DMODEL + d] = (f16)v;
}

// Kernel 0b: f32 -> f16 weight conversion.
__global__ void cvt_f16(const float* __restrict__ src, f16* __restrict__ dst, int n) {
    int i = blockIdx.x * blockDim.x + threadIdx.x;
    if (i < n) dst[i] = (f16)src[i];
}

// ---------------------------------------------------------------------------
// Generic WMMA GEMM: C[M,N] = A16[M,K] * W16[N,K]^T  (f32 accumulate)
//   block tile 64x64, 4 waves, wave w owns M-rows [16w,16w+16), K-step 32.
//   Tiles staged by TDM (wave 0 issues DMA, s_wait_tensorcnt, barrier).
// Epilogue modes:
//   0: store f16 to Out[m*N + n]                                (in-proj -> xz)
//   1: store f32 to Out[m*48 + n] for n < 48                    (x-proj -> x_dbl)
//   2: store f32 to d_out[(b, (rev?256:0)+n, rev? L-1-l : l)]   (out-proj, fused
//      flip + channel-concat + transpose)
// ---------------------------------------------------------------------------
__global__ __launch_bounds__(128)
void gemm_wmma(const f16* __restrict__ A, const f16* __restrict__ W,
               void* __restrict__ Out, int M, int N, int K, int mode, int rev) {
    __shared__ __align__(16) f16 As[64][32];
    __shared__ __align__(16) f16 Bs[64][32];

    const int tid  = threadIdx.x;
    const int wave = tid >> 5;       // wave32
    const int lane = tid & 31;
    const int m0   = blockIdx.x * 64;
    const int n0   = blockIdx.y * 64;

    v8f acc0 = {}; v8f acc1 = {}; v8f acc2 = {}; v8f acc3 = {};

    const int g  = lane >> 4;        // half-lane group
    const int mL = (lane & 15) + wave * 16;
    const int nl = lane & 15;

    const unsigned ldsA = (unsigned)(uintptr_t)&As[0][0];
    const unsigned ldsB = (unsigned)(uintptr_t)&Bs[0][0];

    for (int k0 = 0; k0 < K; k0 += 32) {
        __syncthreads();             // previous iteration's fragments consumed
        if (tid < 32) {              // wave 0 drives the Tensor Data Mover
            tdm_load_tile_f16(A + (size_t)m0 * K + k0, ldsA,
                              /*rows_avail=*/(unsigned)(M - m0),
                              /*tile_rows=*/64u, (unsigned)K);
            tdm_load_tile_f16(W + (size_t)n0 * K + k0, ldsB,
                              /*rows_avail=*/(unsigned)(N - n0),  // OOB rows -> 0
                              /*tile_rows=*/64u, (unsigned)K);
            __builtin_amdgcn_s_wait_tensorcnt(0);
        }
        __syncthreads();             // LDS tiles visible to all waves

        // A fragment: lane holds two contiguous 16B chunks of its M-row
        FragH af;
        af.q[0] = *(const u32x4*)&As[mL][16 * g];
        af.q[1] = *(const u32x4*)&As[mL][16 * g + 8];

        FragH bf;
        bf.q[0] = *(const u32x4*)&Bs[nl +  0][16 * g];
        bf.q[1] = *(const u32x4*)&Bs[nl +  0][16 * g + 8];
        acc0 = __builtin_amdgcn_wmma_f32_16x16x32_f16(false, af.h, false, bf.h,
                                                      (short)0, acc0, false, false);
        bf.q[0] = *(const u32x4*)&Bs[nl + 16][16 * g];
        bf.q[1] = *(const u32x4*)&Bs[nl + 16][16 * g + 8];
        acc1 = __builtin_amdgcn_wmma_f32_16x16x32_f16(false, af.h, false, bf.h,
                                                      (short)0, acc1, false, false);
        bf.q[0] = *(const u32x4*)&Bs[nl + 32][16 * g];
        bf.q[1] = *(const u32x4*)&Bs[nl + 32][16 * g + 8];
        acc2 = __builtin_amdgcn_wmma_f32_16x16x32_f16(false, af.h, false, bf.h,
                                                      (short)0, acc2, false, false);
        bf.q[0] = *(const u32x4*)&Bs[nl + 48][16 * g];
        bf.q[1] = *(const u32x4*)&Bs[nl + 48][16 * g + 8];
        acc3 = __builtin_amdgcn_wmma_f32_16x16x32_f16(false, af.h, false, bf.h,
                                                      (short)0, acc3, false, false);
    }

    // ---- epilogue: C/D layout M = r + 8*g, N = lane%16 (+16*j) ----
    v8f accs[4] = {acc0, acc1, acc2, acc3};
    for (int j = 0; j < 4; ++j) {
        for (int r = 0; r < 8; ++r) {
            const int m = m0 + wave * 16 + g * 8 + r;
            const int n = n0 + j * 16 + nl;
            const float v = accs[j][r];
            if (mode == 0) {
                ((f16*)Out)[(size_t)m * N + n] = (f16)v;
            } else if (mode == 1) {
                if (n < DTRANK + 2 * NSTATE)
                    ((float*)Out)[(size_t)m * 48 + n] = v;
            } else {
                const int b = m / SEQ, l = m % SEQ;
                const int lo = rev ? (SEQ - 1 - l) : l;
                const int c  = (rev ? DMODEL : 0) + n;
                ((float*)Out)[((size_t)b * (2 * DMODEL) + c) * SEQ + lo] = v;
            }
        }
    }
}

// ---------------------------------------------------------------------------
// Kernel 2: causal depthwise conv1d (K=4, left pad 3) + bias + SiLU.
// Reads xi = xz[:, 0:512] (f16), writes xc (f16).
// ---------------------------------------------------------------------------
__global__ void conv_silu(const f16* __restrict__ xz, const float* __restrict__ cw,
                          const float* __restrict__ cb, f16* __restrict__ xc) {
    int idx = blockIdx.x * blockDim.x + threadIdx.x;
    if (idx >= MTOT * DINNER) return;
    const int c = idx % DINNER;
    const int m = idx / DINNER;
    const int l = m % SEQ;
    const int b = m / SEQ;
    float s = cb[c];
#pragma unroll
    for (int k = 0; k < DCONV; ++k) {
        const int ls = l + k - (DCONV - 1);
        if (ls >= 0)
            s += (float)xz[((size_t)(b * SEQ + ls)) * (2 * DINNER) + c] * cw[c * DCONV + k];
    }
    const float sig = 1.f / (1.f + __expf(-s));
    xc[(size_t)m * DINNER + c] = (f16)(s * sig);
}

// ---------------------------------------------------------------------------
// Kernel 3: fused selective scan. One block per batch, one lane per d_inner
// channel; h[16] in registers; delta = softplus(dt . dt_w[d] + dt_b[d]) computed
// in-lane; epilogue fuses (+xc*D) * silu(z) and stores f16 for the out-proj.
// dt/B/C rows staged in LDS in 64-step chunks.
// ---------------------------------------------------------------------------
__global__ __launch_bounds__(DINNER)
void scan_fused(const float* __restrict__ xdbl, const f16* __restrict__ xc16,
                const f16* __restrict__ xz16, const float* __restrict__ dtw,
                const float* __restrict__ dtb, const float* __restrict__ Alog,
                const float* __restrict__ Dp, f16* __restrict__ y16) {
    const int b = blockIdx.x;
    const int d = threadIdx.x;
    __shared__ float sd[64][48];

    float Ar[NSTATE], Wr[DTRANK], h[NSTATE];
#pragma unroll
    for (int n = 0; n < NSTATE; ++n) { Ar[n] = -__expf(Alog[d * NSTATE + n]); h[n] = 0.f; }
#pragma unroll
    for (int r = 0; r < DTRANK; ++r) Wr[r] = dtw[d * DTRANK + r];
    const float bias = dtb[d];
    const float Dd   = Dp[d];

    for (int c0 = 0; c0 < SEQ; c0 += 64) {
        __syncthreads();
        for (int i = d; i < 64 * 48; i += DINNER)
            ((float*)sd)[i] = xdbl[((size_t)b * SEQ + c0) * 48 + i];
        __syncthreads();

        for (int lc = 0; lc < 64; ++lc) {
            const float* row = sd[lc];
            float acc = bias;
#pragma unroll
            for (int r = 0; r < DTRANK; ++r) acc = fmaf(row[r], Wr[r], acc);
            const float delta = (acc > 20.f) ? acc : log1pf(__expf(acc));

            const size_t base = (size_t)(b * SEQ + c0 + lc);
            const float xt = (float)xc16[base * DINNER + d];
            const float zv = (float)xz16[base * (2 * DINNER) + DINNER + d];
            const float dx = delta * xt;

            float y = 0.f;
#pragma unroll
            for (int n = 0; n < NSTATE; ++n) {
                const float dA = __expf(delta * Ar[n]);
                h[n] = dA * h[n] + dx * row[DTRANK + n];
                y    = fmaf(h[n], row[DTRANK + NSTATE + n], y);
            }
            y = (y + xt * Dd) * (zv * (1.f / (1.f + __expf(-zv))));
            y16[base * DINNER + d] = (f16)y;
        }
    }
}

// ---------------------------------------------------------------------------
extern "C" void kernel_launch(void* const* d_in, const int* in_sizes, int n_in,
                              void* d_out, int out_size, void* d_ws, size_t ws_size,
                              hipStream_t stream) {
    (void)in_sizes; (void)n_in; (void)out_size; (void)ws_size;
    const float* x = (const float*)d_in[0];
    // per-direction params: in_w, conv_w, conv_b, xproj_w, dt_w, dt_b, A_log, D, out_w
    const float* P[2][9];
    for (int dir = 0; dir < 2; ++dir)
        for (int i = 0; i < 9; ++i) P[dir][i] = (const float*)d_in[1 + dir * 9 + i];

    char* ws = (char*)d_ws;
    size_t off = 0;
    auto alloc = [&](size_t bytes) -> char* {
        char* p = ws + off;
        off += (bytes + 255) & ~(size_t)255;
        return p;
    };

    f16 *xt16[2], *win16[2], *wxp16[2], *wout16[2], *xz16[2], *xc16[2], *y16[2];
    float* xdbl[2];
    for (int dir = 0; dir < 2; ++dir) {
        xt16[dir]  = (f16*)alloc((size_t)MTOT * DMODEL * sizeof(f16));
        win16[dir] = (f16*)alloc((size_t)(2 * DINNER) * DMODEL * sizeof(f16));
        wxp16[dir] = (f16*)alloc((size_t)48 * DINNER * sizeof(f16));
        wout16[dir]= (f16*)alloc((size_t)DMODEL * DINNER * sizeof(f16));
        xz16[dir]  = (f16*)alloc((size_t)MTOT * (2 * DINNER) * sizeof(f16));
        xdbl[dir]  = (float*)alloc((size_t)MTOT * 48 * sizeof(float));
        xc16[dir]  = (f16*)alloc((size_t)MTOT * DINNER * sizeof(f16));
        y16[dir]   = (f16*)alloc((size_t)MTOT * DINNER * sizeof(f16));
    }

    // --- phase 0: layout + precision prep ---
    {
        const int n = BATCH * DMODEL * SEQ;
        prep_x<<<(n + 255) / 256, 256, 0, stream>>>(x, xt16[0], xt16[1]);
    }
    for (int dir = 0; dir < 2; ++dir) {
        int n;
        n = 2 * DINNER * DMODEL;
        cvt_f16<<<(n + 255) / 256, 256, 0, stream>>>(P[dir][0], win16[dir], n);
        n = 48 * DINNER;
        cvt_f16<<<(n + 255) / 256, 256, 0, stream>>>(P[dir][3], wxp16[dir], n);
        n = DMODEL * DINNER;
        cvt_f16<<<(n + 255) / 256, 256, 0, stream>>>(P[dir][8], wout16[dir], n);
    }

    // --- phases 1..5 per direction ---
    for (int dir = 0; dir < 2; ++dir) {
        // in-projection: (M,256) x (1024,256)^T -> xz16 (M,1024)
        gemm_wmma<<<dim3(MTOT / 64, (2 * DINNER) / 64), 128, 0, stream>>>(
            xt16[dir], win16[dir], xz16[dir], MTOT, 2 * DINNER, DMODEL, 0, dir);

        // depthwise causal conv + SiLU -> xc16
        {
            const int n = MTOT * DINNER;
            conv_silu<<<(n + 255) / 256, 256, 0, stream>>>(
                xz16[dir], P[dir][1], P[dir][2], xc16[dir]);
        }

        // x-projection: (M,512) x (48,512)^T -> xdbl f32 (M,48)
        gemm_wmma<<<dim3(MTOT / 64, 1), 128, 0, stream>>>(
            xc16[dir], wxp16[dir], xdbl[dir], MTOT, 48, DINNER, 1, dir);

        // fused delta/scan/gate -> y16
        scan_fused<<<BATCH, DINNER, 0, stream>>>(
            xdbl[dir], xc16[dir], xz16[dir],
            P[dir][4], P[dir][5], P[dir][6], P[dir][7], y16[dir]);

        // out-projection: (M,512) x (256,512)^T -> d_out (fused flip/concat/T)
        gemm_wmma<<<dim3(MTOT / 64, DMODEL / 64), 128, 0, stream>>>(
            y16[dir], wout16[dir], d_out, MTOT, DMODEL, DINNER, 2, dir);
    }
}
